// MultiHeadAttention_67113158967903
// MI455X (gfx1250) — matrix operand order repaired and checked by reading, not compiled, not run on hardware
//
#include <hip/hip_runtime.h>

// ---------------- problem constants ----------------
#define B_   8
#define T_   1024
#define DM   1024
#define H_   16
#define DH   64
#define BH   (B_ * H_)      // 128
#define QKVN (3 * DM)       // 3072

typedef __bf16 bf16;
typedef __attribute__((ext_vector_type(16))) __bf16 v16bf;
typedef __attribute__((ext_vector_type(8)))  __bf16 v8bf;
typedef __attribute__((ext_vector_type(8)))  float  v8f;
typedef __attribute__((ext_vector_type(4))) unsigned int u32x4;
typedef __attribute__((ext_vector_type(8))) int i32x8;
typedef __attribute__((ext_vector_type(4))) int i32x4;

// ---------------- helpers ----------------
__device__ __forceinline__ unsigned short f2bf(float f) {
  union { float f; unsigned u; } v; v.f = f;
  unsigned r = v.u + 0x7FFFu + ((v.u >> 16) & 1u);   // round-to-nearest-even
  return (unsigned short)(r >> 16);
}
__device__ __forceinline__ float bf2f(unsigned short s) {
  union { float f; unsigned u; } v; v.u = ((unsigned)s) << 16;
  return v.f;
}
__device__ __forceinline__ v8f vzero8() {
  v8f z;
#pragma unroll
  for (int i = 0; i < 8; ++i) z[i] = 0.0f;
  return z;
}

// A-matrix 16x32 bf16 fragment (lane m = lane&15, hk = lane>>4):
// element j -> k = hk*16 + j  => one contiguous 16-elem chunk.
__device__ __forceinline__ v16bf load_a16(const bf16* rowk, int hk) {
  return *(const v16bf*)(rowk + hk * 16);
}
// B-matrix 32x16 bf16 fragment (lane n = lane&15): element j -> k = hk*16 + j.
__device__ __forceinline__ v16bf load_b16(const bf16* rowk, int hk) {
  return *(const v16bf*)(rowk + hk * 16);
}
__device__ __forceinline__ v8f wmma_bf16(v16bf a, v16bf b, v8f c) {
  return __builtin_amdgcn_wmma_f32_16x16x32_bf16(false, a, false, b, (short)0, c,
                                                 false, false);
}

// ---- Tensor Data Mover: 2-D bf16 tile (tile_w elems x tile_h rows) -> LDS ----
// D# per CDNA5 ISA ch.8: group0 {count=1, lds_addr, global_addr, type=2},
// group1 {data_size=2B, tensor_dim0/1 = 1<<30, tile dims, dim0 stride}.
// This toolchain exposes the 6-arg builtin: (g0, g1, g2, g3, extra, cpol).
__device__ __forceinline__ void tdm_load_2d(unsigned lds_off,
                                            unsigned long long gaddr,
                                            unsigned tile_w, unsigned tile_h,
                                            unsigned stride_elems) {
  u32x4 g0;
  g0[0] = 1u;                                                   // count=1 (valid)
  g0[1] = lds_off;                                              // LDS byte addr
  g0[2] = (unsigned)gaddr;                                      // gaddr[31:0]
  g0[3] = (unsigned)((gaddr >> 32) & 0x01FFFFFFu) | (2u << 30); // gaddr[56:32],type=2
  i32x8 g1;
  g1[0] = 1 << 16;                          // workgroup_mask=0, data_size=1 (2B)
  g1[1] = 0;                                // tensor_dim0[15:0] (td0 = 1<<30)
  g1[2] = 0x4000;                           // td0[31:16] | td1[15:0]<<16
  g1[3] = 0x4000 | (int)(tile_w << 16);     // td1[31:16] | tile_dim0
  g1[4] = (int)tile_h;                      // tile_dim1, tile_dim2=0
  g1[5] = (int)stride_elems;                // tensor_dim0_stride[31:0]
  g1[6] = 0;                                // stride[47:32], dim1_stride lo
  g1[7] = 0;
  i32x4 gz;
  gz[0] = 0; gz[1] = 0; gz[2] = 0; gz[3] = 0;
  i32x8 gz8;
#pragma unroll
  for (int i = 0; i < 8; ++i) gz8[i] = 0;
  __builtin_amdgcn_tensor_load_to_lds(g0, g1, gz, gz, gz8, 0);
}
__device__ __forceinline__ unsigned lds_off_of(const void* p) {
  return (unsigned)(unsigned long long)p;   // low 32 bits of flat LDS addr
}

// ---------------- kernel 1: fp32 -> bf16 ----------------
__global__ void cvt_bf16_kernel(const float* __restrict__ src,
                                unsigned short* __restrict__ dst, long long n) {
  long long i = ((long long)blockIdx.x * blockDim.x + threadIdx.x) * 2;
  if (i < n) {
    dst[i]     = f2bf(src[i]);
    dst[i + 1] = f2bf(src[i + 1]);
  }
}

// ---------------- kernel 2: qkv = x @ Wqkv^T + bqkv (bf16 out) ----------------
// grid (M/128, N/128), block 256 (8 waves). W 128x32 tile staged by TDM,
// double-buffered: all 8 waves share it from LDS.
__global__ __launch_bounds__(256)
void gemm_qkv_kernel(const unsigned short* __restrict__ A_,
                     const unsigned short* __restrict__ W_,
                     const float* __restrict__ bias,
                     unsigned short* __restrict__ C_) {
  const bf16* A = (const bf16*)A_;
  int wave = threadIdx.x >> 5, lane = threadIdx.x & 31;
  int hk = lane >> 4, ln = lane & 15;
  int m0 = blockIdx.x * 128 + wave * 16;
  int n0 = blockIdx.y * 128;

  __shared__ __align__(128) unsigned short ldsW[2][128 * 32];
  unsigned loff[2] = { lds_off_of(&ldsW[0][0]), lds_off_of(&ldsW[1][0]) };
  unsigned long long wg = (unsigned long long)(const void*)(W_ + (size_t)n0 * DM);

  v8f acc[8];
#pragma unroll
  for (int s = 0; s < 8; ++s) acc[s] = vzero8();

  const bf16* arow = A + (size_t)(m0 + ln) * DM;
  const int nIter = DM / 32;
  if (threadIdx.x < 32) tdm_load_2d(loff[0], wg, 32, 128, DM);

  for (int i = 0; i < nIter; ++i) {
    if (threadIdx.x < 32) {
      if (i + 1 < nIter) {
        tdm_load_2d(loff[(i + 1) & 1], wg + (size_t)(i + 1) * 32 * 2, 32, 128, DM);
        __builtin_amdgcn_s_wait_tensorcnt(1);   // tile i has landed
      } else {
        __builtin_amdgcn_s_wait_tensorcnt(0);
      }
    }
    __syncthreads();                            // publish tile i
    v16bf a = load_a16(arow + i * 32, hk);
    const bf16* wl = (const bf16*)ldsW[i & 1];
#pragma unroll
    for (int s = 0; s < 8; ++s) {
      v16bf b = *(const v16bf*)(wl + (size_t)(s * 16 + ln) * 32 + hk * 16);
      acc[s] = wmma_bf16(a, b, acc[s]);
    }
    __syncthreads();                            // protect buffer before overwrite
  }
#pragma unroll
  for (int s = 0; s < 8; ++s) {
    int n = n0 + s * 16 + ln;
    float bv = bias[n];
#pragma unroll
    for (int r = 0; r < 8; ++r) {
      int m = m0 + r + 8 * hk;
      C_[(size_t)m * QKVN + n] = f2bf(acc[s][r] + bv);
    }
  }
}

// ---------------- kernel 3: vT[bh, d, t] = V[b, t, h, d] ----------------
__global__ void vtrans_kernel(const unsigned short* __restrict__ qkv,
                              unsigned short* __restrict__ vT) {
  long long i = (long long)blockIdx.x * blockDim.x + threadIdx.x;
  int t = (int)(i & (T_ - 1));
  long long j = i >> 10;
  int d = (int)(j & (DH - 1));
  int bh = (int)(j >> 6);
  int b = bh >> 4, h = bh & 15;
  vT[i] = qkv[(size_t)(b * T_ + t) * QKVN + 2 * DM + h * DH + d];
}

// ---------------- kernel 4: rel[t, bh, s] = sum_d q[bh,t,d]*pos[t,s,d] -------
// grid (T/256 s-blocks = 4, T), block 256. pos tile 256x64 (32KB) via one TDM,
// shared by all 8 waves from LDS.
__global__ __launch_bounds__(256)
void relpos_kernel(const unsigned short* __restrict__ qkv_,
                   const unsigned short* __restrict__ pos_,
                   unsigned short* __restrict__ rel) {
  const bf16* qkv = (const bf16*)qkv_;
  int t = blockIdx.y;
  int s0 = blockIdx.x * 256;
  int wave = threadIdx.x >> 5, lane = threadIdx.x & 31;
  int hk = lane >> 4, ln = lane & 15;

  __shared__ __align__(128) unsigned short ldsP[256 * 64];
  if (threadIdx.x < 32) {
    unsigned long long g =
        (unsigned long long)(const void*)(pos_ + ((size_t)t * T_ + s0) * DH);
    tdm_load_2d(lds_off_of(&ldsP[0]), g, 64, 256, DH);
    __builtin_amdgcn_s_wait_tensorcnt(0);
  }

  int arow_bh = wave * 16 + ln;                     // A-matrix row = bh index
  int b = arow_bh >> 4, h = arow_bh & 15;
  const bf16* qrow = qkv + (size_t)(b * T_ + t) * QKVN + h * DH;
  v16bf a0 = load_a16(qrow, hk);                    // d = 0..31
  v16bf a1 = load_a16(qrow + 32, hk);               // d = 32..63

  __syncthreads();                                  // pos tile visible
  const bf16* pl = (const bf16*)ldsP;

#pragma unroll 4
  for (int sub = 0; sub < 16; ++sub) {
    const bf16* prow = pl + (size_t)(sub * 16 + ln) * DH;
    v16bf b0 = load_b16(prow, hk);
    v16bf b1 = load_b16(prow + 32, hk);
    v8f c = vzero8();
    c = wmma_bf16(a0, b0, c);
    c = wmma_bf16(a1, b1, c);
#pragma unroll
    for (int r = 0; r < 8; ++r) {
      int bh = wave * 16 + r + 8 * hk;
      rel[((size_t)t * BH + bh) * T_ + s0 + sub * 16 + ln] = f2bf(c[r]);
    }
  }
}

// ---------------- kernel 5: flash attention (1 wave per (b,h,16 t-rows)) -----
__global__ __launch_bounds__(32)
void flash_kernel(const unsigned short* __restrict__ qkv_,
                  const unsigned short* __restrict__ vT_,
                  const unsigned short* __restrict__ rel_,
                  unsigned short* __restrict__ ctx) {
  const bf16* qkv = (const bf16*)qkv_;
  const bf16* vT  = (const bf16*)vT_;
  int tt = blockIdx.x & 63;            // T/16 tiles
  int bh = blockIdx.x >> 6;
  int b = bh >> 4, h = bh & 15;
  int t0 = tt * 16;
  int lane = threadIdx.x;
  int hk = lane >> 4, ln = lane & 15;

  // Q fragments stay in registers for the whole s-loop
  const bf16* qrow = qkv + (size_t)(b * T_ + t0 + ln) * QKVN + h * DH;
  v16bf qa0 = load_a16(qrow, hk);
  v16bf qa1 = load_a16(qrow + 32, hk);

  const bf16* kb_base = qkv + (size_t)(b * T_) * QKVN + DM + h * DH;
  const bf16* v_base  = vT + (size_t)bh * DH * T_;
  const unsigned short* relrow = rel_ + ((size_t)t0 * BH + bh) * T_;

  float mr[8], lr[8];
  v8f acc[4];
#pragma unroll
  for (int r = 0; r < 8; ++r) { mr[r] = -3.0e38f; lr[r] = 0.0f; }
#pragma unroll
  for (int d = 0; d < 4; ++d) acc[d] = vzero8();

  __shared__ __align__(64) unsigned short pls[16 * 32];  // P tile (C->A relayout)

  for (int s0 = 0; s0 < T_; s0 += 32) {
    // ---- scores: two 16x16 subtiles, K=64 over d ----
    v8f sc[2];
#pragma unroll
    for (int sub = 0; sub < 2; ++sub) {
      const bf16* krow = kb_base + (size_t)(s0 + sub * 16 + ln) * QKVN;
      v16bf kb0 = load_b16(krow, hk);
      v16bf kb1 = load_b16(krow + 32, hk);
      v8f c = vzero8();
      c = wmma_bf16(qa0, kb0, c);
      c = wmma_bf16(qa1, kb1, c);
#pragma unroll
      for (int r = 0; r < 8; ++r) {
        int m = r + 8 * hk;
        float rv = bf2f(relrow[(size_t)m * (BH * T_) + s0 + sub * 16 + ln]);
        c[r] = (c[r] + rv) * 0.125f;   // (qk + rel) / sqrt(dh)
      }
      sc[sub] = c;
    }
    // ---- online softmax (reduce over n within each 16-lane half) ----
    float rmax[8];
#pragma unroll
    for (int r = 0; r < 8; ++r) rmax[r] = fmaxf(sc[0][r], sc[1][r]);
#pragma unroll
    for (int mask = 1; mask < 16; mask <<= 1)
#pragma unroll
      for (int r = 0; r < 8; ++r)
        rmax[r] = fmaxf(rmax[r], __shfl_xor(rmax[r], mask, 32));
    float al[8];
#pragma unroll
    for (int r = 0; r < 8; ++r) {
      float mn = fmaxf(mr[r], rmax[r]);
      al[r] = __expf(mr[r] - mn);
      mr[r] = mn;
    }
#pragma unroll
    for (int sub = 0; sub < 2; ++sub)
#pragma unroll
      for (int r = 0; r < 8; ++r) sc[sub][r] = __expf(sc[sub][r] - mr[r]);
    float rs[8];
#pragma unroll
    for (int r = 0; r < 8; ++r) rs[r] = sc[0][r] + sc[1][r];
#pragma unroll
    for (int mask = 1; mask < 16; mask <<= 1)
#pragma unroll
      for (int r = 0; r < 8; ++r) rs[r] += __shfl_xor(rs[r], mask, 32);
#pragma unroll
    for (int r = 0; r < 8; ++r) lr[r] = lr[r] * al[r] + rs[r];
#pragma unroll
    for (int d = 0; d < 4; ++d)
#pragma unroll
      for (int r = 0; r < 8; ++r) acc[d][r] *= al[r];

    // ---- P: C-layout -> A-fragment layout via LDS ----
#pragma unroll
    for (int sub = 0; sub < 2; ++sub)
#pragma unroll
      for (int r = 0; r < 8; ++r) {
        int m = r + 8 * hk;
        pls[m * 32 + sub * 16 + ln] = f2bf(sc[sub][r]);
      }
    asm volatile("s_wait_dscnt 0x0" ::: "memory");  // same-wave LDS RAW fence
    const bf16* prow = (const bf16*)&pls[ln * 32];
    v16bf pa = load_a16(prow, hk);                  // P 16x32 A fragment

    // ---- acc += P @ V (V transposed => contiguous B fragments) ----
#pragma unroll
    for (int d = 0; d < 4; ++d) {
      const bf16* vrow = v_base + (size_t)(d * 16 + ln) * T_ + s0;
      v16bf vb = load_b16(vrow, hk);
      acc[d] = wmma_bf16(pa, vb, acc[d]);
    }
  }

  // ---- epilogue: ctx[b,t,h,d] = acc / l ----
#pragma unroll
  for (int d = 0; d < 4; ++d)
#pragma unroll
    for (int r = 0; r < 8; ++r) {
      int m = r + 8 * hk;
      float o = acc[d][r] / lr[r];
      ctx[(size_t)(b * T_ + t0 + m) * DM + h * DH + d * 16 + ln] = f2bf(o);
    }
}

// ---------------- kernel 6: out = ctx @ Wout^T + bout (fp32 out) -------------
__global__ __launch_bounds__(256)
void gemm_out_kernel(const unsigned short* __restrict__ A_,
                     const unsigned short* __restrict__ W_,
                     const float* __restrict__ bias,
                     float* __restrict__ O) {
  const bf16* A = (const bf16*)A_;
  int wave = threadIdx.x >> 5, lane = threadIdx.x & 31;
  int hk = lane >> 4, ln = lane & 15;
  int m0 = blockIdx.x * 128 + wave * 16;
  int n0 = blockIdx.y * 128;

  __shared__ __align__(128) unsigned short ldsW[2][128 * 32];
  unsigned loff[2] = { lds_off_of(&ldsW[0][0]), lds_off_of(&ldsW[1][0]) };
  unsigned long long wg = (unsigned long long)(const void*)(W_ + (size_t)n0 * DM);

  v8f acc[8];
#pragma unroll
  for (int s = 0; s < 8; ++s) acc[s] = vzero8();

  const bf16* arow = A + (size_t)(m0 + ln) * DM;
  const int nIter = DM / 32;
  if (threadIdx.x < 32) tdm_load_2d(loff[0], wg, 32, 128, DM);

  for (int i = 0; i < nIter; ++i) {
    if (threadIdx.x < 32) {
      if (i + 1 < nIter) {
        tdm_load_2d(loff[(i + 1) & 1], wg + (size_t)(i + 1) * 32 * 2, 32, 128, DM);
        __builtin_amdgcn_s_wait_tensorcnt(1);
      } else {
        __builtin_amdgcn_s_wait_tensorcnt(0);
      }
    }
    __syncthreads();
    v16bf a = load_a16(arow + i * 32, hk);
    const bf16* wl = (const bf16*)ldsW[i & 1];
#pragma unroll
    for (int s = 0; s < 8; ++s) {
      v16bf b = *(const v16bf*)(wl + (size_t)(s * 16 + ln) * 32 + hk * 16);
      acc[s] = wmma_bf16(a, b, acc[s]);
    }
    __syncthreads();
  }
#pragma unroll
  for (int s = 0; s < 8; ++s) {
    int n = n0 + s * 16 + ln;
    float bv = bias[n];
#pragma unroll
    for (int r = 0; r < 8; ++r) {
      int m = m0 + r + 8 * hk;
      O[(size_t)m * DM + n] = acc[s][r] + bv;
    }
  }
}

// ---------------- host-side launch ----------------
extern "C" void kernel_launch(void* const* d_in, const int* in_sizes, int n_in,
                              void* d_out, int out_size, void* d_ws, size_t ws_size,
                              hipStream_t stream) {
  const float* x    = (const float*)d_in[0];   // (B,T,DM)
  const float* pos  = (const float*)d_in[1];   // (T,T,DH)
  const float* Wqkv = (const float*)d_in[2];   // (3DM,DM)
  const float* bqkv = (const float*)d_in[3];   // (3DM)
  const float* Wout = (const float*)d_in[4];   // (DM,DM)
  const float* bout = (const float*)d_in[5];   // (DM)
  float* out = (float*)d_out;

  // workspace carve (256B aligned)
  char* w = (char*)d_ws;
  auto carve = [&](size_t bytes) {
    void* p = (void*)w;
    w += (bytes + 255) & ~(size_t)255;
    return p;
  };
  const long long nx   = (long long)B_ * T_ * DM;        // 8388608
  const long long npos = (long long)T_ * T_ * DH;        // 67108864
  const long long nwq  = (long long)QKVN * DM;           // 3145728
  const long long nwo  = (long long)DM * DM;             // 1048576
  unsigned short* xb    = (unsigned short*)carve(nx * 2);
  unsigned short* wqkvb = (unsigned short*)carve(nwq * 2);
  unsigned short* woutb = (unsigned short*)carve(nwo * 2);
  unsigned short* posb  = (unsigned short*)carve(npos * 2);
  unsigned short* qkvb  = (unsigned short*)carve((long long)B_ * T_ * QKVN * 2);
  unsigned short* relb  = (unsigned short*)carve((long long)T_ * BH * T_ * 2);
  unsigned short* ctxb  = (unsigned short*)carve(nx * 2);
  unsigned short* vtb   = (unsigned short*)carve(nx * 2);

  // 1) conversions to bf16
  cvt_bf16_kernel<<<(unsigned)(nx   / 512), 256, 0, stream>>>(x,    xb,    nx);
  cvt_bf16_kernel<<<(unsigned)(nwq  / 512), 256, 0, stream>>>(Wqkv, wqkvb, nwq);
  cvt_bf16_kernel<<<(unsigned)(nwo  / 512), 256, 0, stream>>>(Wout, woutb, nwo);
  cvt_bf16_kernel<<<(unsigned)(npos / 512), 256, 0, stream>>>(pos,  posb,  npos);

  // 2) QKV projection: (8192 x 3072) = x @ Wqkv^T + bqkv
  gemm_qkv_kernel<<<dim3(B_ * T_ / 128, QKVN / 128), 256, 0, stream>>>(
      xb, wqkvb, bqkv, qkvb);

  // 3) V transpose -> (bh, d, t)
  vtrans_kernel<<<(unsigned)(nx / 256), 256, 0, stream>>>(qkvb, vtb);

  // 4) relative-position scores: rel[t, bh, s]
  relpos_kernel<<<dim3(T_ / 256, T_), 256, 0, stream>>>(qkvb, posb, relb);

  // 5) flash attention -> ctx bf16 (B,T,DM)
  flash_kernel<<<(unsigned)(BH * (T_ / 16)), 32, 0, stream>>>(qkvb, vtb, relb, ctxb);

  // 6) output projection -> d_out fp32
  gemm_out_kernel<<<dim3(B_ * T_ / 128, DM / 128), 256, 0, stream>>>(
      ctxb, woutb, bout, out);
}
